// SCOFFCell_71313636983345
// MI455X (gfx1250) — compile-verified, matrix operand order checked by reading, not run on hardware
//
#include <hip/hip_runtime.h>
#include <hip/hip_bf16.h>

// ---------------------------------------------------------------------------
// SCOFF cell forward for MI455X (gfx1250, wave32, WMMA).
//
// Roofline: ~20 G MACs (40 GFLOP) vs ~85 MB HBM traffic (~3.7 us at 23.3 TB/s)
// -> matrix-core bound. Strategy: bf16 v_wmma_f32_16x16x32_bf16 for every
// K=128 GEMM (f32 accumulate); weights (~2.4 MB, L2-resident on 192 MB L2)
// are pre-converted to bf16 and pre-swizzled into the CDNA5 B-fragment layout
// (ISA 7.12.2: lane = N column, 16 K slots with half-wave interleave) so each
// wave fetches a B fragment with one contiguous 32B load (2x global_load_b128).
// One workgroup (8 wave32) handles 2 batch rows = 16 (b,m) rows, exactly
// filling the WMMA M=16 tile for the dominant GRU/projection GEMMs.
// Activations are staged into LDS with GLOBAL_LOAD_ASYNC_TO_LDS_B32 (ASYNCcnt)
// and converted f32->bf16 in place; h_read/w_key are WMMA'd as well.
// ---------------------------------------------------------------------------

typedef __attribute__((ext_vector_type(16))) __bf16 v16bf;
typedef __attribute__((ext_vector_type(8)))  float  v8f;

constexpr int  Bsz = 4096, Mb = 8, HSS = 128, HID = Mb * HSS;
constexpr int  NIN = 8, DIN = 128, TOPK = 4, Tt = 4;
constexpr int  IH = 4, KDIM = 64, VDIM = 128;
constexpr int  CH = 4, CDK = 32, CDV = 32;

constexpr size_t OFF_HX = 0;
constexpr size_t OFF_MF = (size_t)Bsz * HID;
constexpr size_t OFF_BM = OFF_MF + (size_t)Bsz * HID;
constexpr size_t OFF_TA = OFF_BM + (size_t)Bsz * Mb;

constexpr size_t PW_WQ  = 0;
constexpr size_t PW_WK  = PW_WQ  + 128 * 256;
constexpr size_t PW_WV  = PW_WK  + 128 * 256;
constexpr size_t PW_WIH = PW_WV  + 128 * 512;
constexpr size_t PW_WHH = PW_WIH + 128 * 1536;
constexpr size_t PW_WQC = PW_WHH + 128 * 1536;
constexpr size_t PW_WKC = PW_WQC + 128 * 128;
constexpr size_t PW_WVC = PW_WKC + 128 * 128;
constexpr size_t PW_WFC = PW_WVC + 128 * 128;
constexpr size_t PW_WG  = PW_WFC + 128 * 128;
constexpr size_t PW_WRD = PW_WG  + 128 * 128;   // W_read  128x16
constexpr size_t PW_WWR = PW_WRD + 128 * 16;    // W_write 128x64 (t-blocked)

__device__ __forceinline__ unsigned short f2bf(float x) {
  unsigned u = __builtin_bit_cast(unsigned, x);
  u += 0x7FFFu + ((u >> 16) & 1u);                 // round-to-nearest-even
  return (unsigned short)(u >> 16);
}
__device__ __forceinline__ float bf2f(unsigned short s) {
  return __builtin_bit_cast(float, (unsigned)s << 16);
}
__device__ __forceinline__ float sigm(float x) { return 1.0f / (1.0f + __expf(-x)); }

__device__ __forceinline__ v8f zero8() {
  v8f z;
#pragma unroll
  for (int i = 0; i < 8; ++i) z[i] = 0.0f;
  return z;
}

// Async Global -> LDS copy of one dword per lane (tracked by ASYNCcnt).
__device__ __forceinline__ void async_g2l_b32(void* lds_ptr, const float* gptr) {
  unsigned lds = (unsigned)(unsigned long long)lds_ptr;   // addr[31:0] = LDS offset
  asm volatile("global_load_async_to_lds_b32 %0, %1, off"
               :: "v"(lds), "v"(gptr) : "memory");
}
__device__ __forceinline__ void wait_asynccnt0() {
  asm volatile("s_wait_asynccnt 0" ::: "memory");
}

// A fragment (16x32 bf16 slice of a [16][128] LDS matrix), ISA 7.12.2 layout.
__device__ __forceinline__ v16bf fragA(const unsigned short* A, int k0, int lane) {
  const int half = (lane >> 4) & 1;
  const int m    = lane & 15;
  const unsigned short* row = A + m * 128 + k0 + half * 8;
  v16bf a;
#pragma unroll
  for (int j = 0; j < 8; ++j) a[j]     = __builtin_bit_cast(__bf16, row[j]);
#pragma unroll
  for (int j = 0; j < 8; ++j) a[8 + j] = __builtin_bit_cast(__bf16, row[16 + j]);
  return a;
}

// B fragment: pre-swizzled as [ktile][ntile][lane][16] bf16 -> one 32B load.
__device__ __forceinline__ v16bf fragB(const unsigned short* W, int wnt, int kt,
                                       int nt, int lane) {
  const unsigned short* p = W + (((size_t)kt * wnt + nt) * 32 + lane) * 16;
  return *reinterpret_cast<const v16bf*>(p);
}

// C := A1(16x128) @ W1[:,16-col tile nt1] (+ A2 @ W2[:,nt2]) (+ bias).
__device__ __forceinline__ void mm16(const unsigned short* A1, const unsigned short* W1,
                                     int nt1, int wnt1,
                                     const unsigned short* A2, const unsigned short* W2,
                                     int nt2, int wnt2,
                                     float* C, int ldc, int c0,
                                     const float* b1, const float* b2, int lane) {
  v8f acc = zero8();
#pragma unroll
  for (int kt = 0; kt < 4; ++kt)
    acc = __builtin_amdgcn_wmma_f32_16x16x32_bf16(false, fragA(A1, kt * 32, lane),
                                                  false, fragB(W1, wnt1, kt, nt1, lane),
                                                  (short)0, acc, false, false);
  if (A2) {
#pragma unroll
    for (int kt = 0; kt < 4; ++kt)
      acc = __builtin_amdgcn_wmma_f32_16x16x32_bf16(false, fragA(A2, kt * 32, lane),
                                                    false, fragB(W2, wnt2, kt, nt2, lane),
                                                    (short)0, acc, false, false);
  }
  const int half = (lane >> 4) & 1, n = lane & 15;
  if (b1 || b2) {                                   // avoid dead +0.0 adds
    float bv = (b1 ? b1[n] : 0.0f) + (b2 ? b2[n] : 0.0f);
#pragma unroll
    for (int r = 0; r < 8; ++r)
      C[(size_t)(half * 8 + r) * ldc + c0 + n] = acc[r] + bv;
  } else {
#pragma unroll
    for (int r = 0; r < 8; ++r)
      C[(size_t)(half * 8 + r) * ldc + c0 + n] = acc[r];
  }
}

// Dynamic LDS plan (~118 KB, phase-unioned; gfx1250 allows 320 KB/workgroup).
struct SMem {
  unsigned short Hs[16][128];
  unsigned short Xs[2][16][128];
  unsigned short Xg[16][128];
  unsigned short hnexts[4][16][128];
  unsigned short Hn[16][128];
  unsigned short outc[16][128];
  float Hnf[16][128];
  float probs[2][8][9];
  float notnull[16];
  float maskv[16];
  float hread[16][16];
  float wkey[16][4][16];
  float attw[16][4];
  union {
    struct { float stageH[2048]; float stageX[2048]; }                       p0;
    struct { float Qs[16][256]; float Ks[2][16][256]; float Vs[2][16][128]; } p1;
    struct { float Grz[16][256]; float Gin[16][128]; float Ghn[16][128]; }   p3;
    struct { float qc[16][128]; float kc[16][128]; float vc[16][128];
             float attn[2][4][8][8]; }                                       p5;
    struct { float g1[16][128]; float g2[16][128]; }                         p6;
  } u;
};

// Weight packing: f32 -> bf16 WMMA-B fragments.
// mode 0: src is (K,N) row-major.  mode 1: src is (N,K) row-major (transposed).
// mode 2: W_write layout (T,128,16): B[k][t*16+j] = src[t*2048 + k*16 + j].
__global__ void pack_weight(const float* __restrict__ src, unsigned short* __restrict__ dst,
                            int K, int N, int mode) {
  int o = blockIdx.x * blockDim.x + threadIdx.x;
  int total = K * N;
  if (o >= total) return;
  int ntiles = N >> 4;
  int j    = o & 15;
  int lane = (o >> 4) & 31;
  int tile = o >> 9;
  int nt   = tile % ntiles;
  int kt   = tile / ntiles;
  int n    = nt * 16 + (lane & 15);
  int half = lane >> 4;
  int kl   = (j < 8) ? (half * 8 + j) : (8 + half * 8 + j);
  int k    = kt * 32 + kl;
  float v;
  if (mode == 1)      v = src[(size_t)n * K + k];
  else if (mode == 2) v = src[(size_t)(n >> 4) * 2048 + (size_t)k * 16 + (n & 15)];
  else                v = src[(size_t)k * N + n];
  dst[o] = f2bf(v);
}

__global__ __launch_bounds__(256) void scoff_cell_kernel(
    const float* __restrict__ inp, const float* __restrict__ hx,
    const float* __restrict__ b_ih, const float* __restrict__ b_hh,
    const float* __restrict__ bq_c, const float* __restrict__ bk_c,
    const float* __restrict__ bv_c, const float* __restrict__ b_fc,
    const float* __restrict__ b_gate,
    const unsigned short* __restrict__ Wq_p,  const unsigned short* __restrict__ Wk_p,
    const unsigned short* __restrict__ Wv_p,  const unsigned short* __restrict__ Wih_p,
    const unsigned short* __restrict__ Whh_p, const unsigned short* __restrict__ Wqc_p,
    const unsigned short* __restrict__ Wkc_p, const unsigned short* __restrict__ Wvc_p,
    const unsigned short* __restrict__ Wfc_p, const unsigned short* __restrict__ Wg_p,
    const unsigned short* __restrict__ Wrd_p, const unsigned short* __restrict__ Wwr_p,
    float* __restrict__ out) {
  extern __shared__ char smem_raw[];
  SMem& S = *reinterpret_cast<SMem*>(smem_raw);

  const int tid  = threadIdx.x;
  const int wid  = tid >> 5;
  const int lane = tid & 31;
  const int b0   = blockIdx.x * 2;

  if (tid == 0) { __builtin_prefetch(Wih_p, 0, 1); __builtin_prefetch(Whh_p, 0, 1); }

  // phase 0: async-stage activations (global f32 -> LDS), then convert to bf16.
  for (int idx = tid; idx < 2048; idx += 256) {
    int r = idx >> 7, d = idx & 127;
    async_g2l_b32(&S.u.p0.stageH[idx],
                  hx + (size_t)(b0 + (r >> 3)) * HID + (r & 7) * HSS + d);
  }
  for (int idx = tid; idx < 2048; idx += 256) {
    int i = idx >> 10, n = (idx >> 7) & 7, d = idx & 127;
    async_g2l_b32(&S.u.p0.stageX[idx],
                  inp + ((size_t)(b0 + i) * NIN + n) * DIN + d);
  }
  wait_asynccnt0();                                  // my copies landed in LDS
  for (int idx = tid; idx < 2048; idx += 256) {      // convert the same elems
    int r = idx >> 7, d = idx & 127;
    S.Hs[r][d] = f2bf(S.u.p0.stageH[idx]);
  }
  for (int idx = tid; idx < 2048; idx += 256) {
    int i = idx >> 10, n = (idx >> 7) & 7, d = idx & 127;
    S.Xs[i][n][d] = f2bf(S.u.p0.stageX[idx]);
  }
  for (int idx = tid; idx < 2048; idx += 256) {      // null token + pad rows = 0
    int i = idx >> 10, n = 8 + ((idx >> 7) & 7), d = idx & 127;
    S.Xs[i][n][d] = 0;
  }
  __syncthreads();

  // phase 1 (WMMA): Q = H@Wq ; K_i = Xn_i@Wk ; V_i = mean_IH(Xn_i@Wv)
  for (int task = wid; task < 48; task += 8) {
    if (task < 16) {
      mm16(&S.Hs[0][0], Wq_p, task, 16, nullptr, nullptr, 0, 0,
           &S.u.p1.Qs[0][0], 256, task * 16, nullptr, nullptr, lane);
    } else {
      int tt = task - 16, i = tt >> 4, nt = tt & 15;
      mm16(&S.Xs[i][0][0], Wk_p, nt, 16, nullptr, nullptr, 0, 0,
           &S.u.p1.Ks[i][0][0], 256, nt * 16, nullptr, nullptr, lane);
    }
  }
  for (int task = wid; task < 16; task += 8) {
    int i = task >> 3, dt = task & 7;
    v8f acc = zero8();
#pragma unroll
    for (int h = 0; h < IH; ++h)
#pragma unroll
      for (int kt = 0; kt < 4; ++kt)
        acc = __builtin_amdgcn_wmma_f32_16x16x32_bf16(
            false, fragA(&S.Xs[i][0][0], kt * 32, lane),
            false, fragB(Wv_p, 32, kt, h * 8 + dt, lane),
            (short)0, acc, false, false);
    const int half = (lane >> 4) & 1, n = lane & 15;
#pragma unroll
    for (int r = 0; r < 8; ++r)
      S.u.p1.Vs[i][half * 8 + r][dt * 16 + n] = 0.25f * acc[r];
  }
  __syncthreads();

  // phase 2 (VALU): scores, softmax, top-k mask, masked inp_use -> Xg
  if (tid < 144) {
    int bl = tid / 72, rem = tid % 72, m = rem / 9, n = rem % 9;
    float s = 0.0f;
    for (int c = 0; c < 256; ++c)
      s += S.u.p1.Qs[bl * 8 + m][c] * S.u.p1.Ks[bl][n][c];
    S.probs[bl][m][n] = s * (0.125f * 0.25f);        // /sqrt(64), mean over IH
  }
  __syncthreads();
  if (tid < 16) {
    int bl = tid >> 3, m = tid & 7;
    float mx = -1e30f;
    for (int n = 0; n < 9; ++n) mx = fmaxf(mx, S.probs[bl][m][n]);
    float den = 0.0f;
    for (int n = 0; n < 9; ++n) { float e = expf(S.probs[bl][m][n] - mx);
                                  S.probs[bl][m][n] = e; den += e; }
    float inv = 1.0f / den;
    for (int n = 0; n < 9; ++n) S.probs[bl][m][n] *= inv;
    S.notnull[tid] = 1.0f - S.probs[bl][m][8];
  }
  __syncthreads();
  if (tid < 2) {
    float v[8], mk[8];
    for (int m = 0; m < 8; ++m) { v[m] = S.notnull[tid * 8 + m]; mk[m] = 0.0f; }
    for (int s = 0; s < TOPK; ++s) {
      int best = 0; float bv = v[0];
      for (int m = 1; m < 8; ++m) if (v[m] > bv) { bv = v[m]; best = m; }
      mk[best] = 1.0f; v[best] = -1e30f;
    }
    for (int m = 0; m < 8; ++m) S.maskv[tid * 8 + m] = mk[m];
  }
  __syncthreads();
  for (int idx = tid; idx < 2048; idx += 256) {
    int r = idx >> 7, d = idx & 127, bl = r >> 3, m = r & 7;
    float s = 0.0f;
    for (int n = 0; n < 9; ++n) s += S.probs[bl][m][n] * S.u.p1.Vs[bl][n][d];
    S.Xg[r][d] = f2bf(s * S.maskv[r]);
  }
  __syncthreads();

  // phase 3 (WMMA+VALU): shared GRU over T=4
  for (int t = 0; t < Tt; ++t) {
    for (int task = wid; task < 24; task += 8) {
      if (task < 16) {
        mm16(&S.Xg[0][0], Wih_p, t * 24 + task, 96,
             &S.Hs[0][0], Whh_p, t * 24 + task, 96,
             &S.u.p3.Grz[0][0], 256, task * 16,
             b_ih + t * 384 + task * 16, b_hh + t * 384 + task * 16, lane);
      } else {
        int nt = task - 16;
        mm16(&S.Xg[0][0], Wih_p, t * 24 + 16 + nt, 96, nullptr, nullptr, 0, 0,
             &S.u.p3.Gin[0][0], 128, nt * 16,
             b_ih + t * 384 + 256 + nt * 16, nullptr, lane);
        mm16(&S.Hs[0][0], Whh_p, t * 24 + 16 + nt, 96, nullptr, nullptr, 0, 0,
             &S.u.p3.Ghn[0][0], 128, nt * 16,
             b_hh + t * 384 + 256 + nt * 16, nullptr, lane);
      }
    }
    __syncthreads();
    for (int idx = tid; idx < 2048; idx += 256) {
      int r = idx >> 7, d = idx & 127;
      float rr = sigm(S.u.p3.Grz[r][d]);
      float zz = sigm(S.u.p3.Grz[r][128 + d]);
      float nn = tanhf(S.u.p3.Gin[r][d] + rr * S.u.p3.Ghn[r][d]);
      float ho = bf2f(S.Hs[r][d]);
      S.hnexts[t][r][d] = f2bf((1.0f - zz) * nn + zz * ho);
    }
    __syncthreads();
  }

  // phase 4 (WMMA+VALU): temporal attention, hnew
  for (int task = wid; task < 5; task += 8) {        // h_read + 4x w_key, WMMA
    if (task == 0)
      mm16(&S.Hs[0][0], Wrd_p, 0, 1, nullptr, nullptr, 0, 0,
           &S.hread[0][0], 16, 0, nullptr, nullptr, lane);
    else {
      int t = task - 1;
      mm16(&S.hnexts[t][0][0], Wwr_p, t, 4, nullptr, nullptr, 0, 0,
           &S.wkey[0][0][0], 64, t * 16, nullptr, nullptr, lane);
    }
  }
  __syncthreads();
  if (tid < 16) {
    int r = tid;
    float lg[4]; float mx = -1e30f;
    for (int t = 0; t < 4; ++t) {
      float s = 0.0f;
      for (int j = 0; j < 16; ++j) s += S.hread[r][j] * S.wkey[r][t][j];
      lg[t] = s * 0.25f; mx = fmaxf(mx, lg[t]);
    }
    float den = 0.0f;
    for (int t = 0; t < 4; ++t) { lg[t] = expf(lg[t] - mx); den += lg[t]; }
    for (int t = 0; t < 4; ++t) {
      float a = lg[t] / den;
      S.attw[r][t] = a;
      out[OFF_TA + (size_t)(b0 * 8 + r) * 4 + t] = a;
    }
  }
  __syncthreads();
  for (int idx = tid; idx < 2048; idx += 256) {
    int r = idx >> 7, d = idx & 127;
    float s = 0.0f;
    for (int t = 0; t < 4; ++t) s += S.attw[r][t] * bf2f(S.hnexts[t][r][d]);
    S.Hnf[r][d] = s;
    S.Hn[r][d]  = f2bf(s);
  }
  __syncthreads();

  // phase 5 (WMMA+VALU): communication attention
  for (int task = wid; task < 24; task += 8) {
    int mat = task >> 3, nt = task & 7;
    if (mat == 0)
      mm16(&S.Hn[0][0], Wqc_p, nt, 8, nullptr, nullptr, 0, 0,
           &S.u.p5.qc[0][0], 128, nt * 16, bq_c + nt * 16, nullptr, lane);
    else if (mat == 1)
      mm16(&S.Hn[0][0], Wkc_p, nt, 8, nullptr, nullptr, 0, 0,
           &S.u.p5.kc[0][0], 128, nt * 16, bk_c + nt * 16, nullptr, lane);
    else
      mm16(&S.Hn[0][0], Wvc_p, nt, 8, nullptr, nullptr, 0, 0,
           &S.u.p5.vc[0][0], 128, nt * 16, bv_c + nt * 16, nullptr, lane);
  }
  __syncthreads();
  for (int idx = tid; idx < 512; idx += 256) {
    int bl = idx >> 8, h = (idx >> 6) & 3, m = (idx >> 3) & 7, n = idx & 7;
    float s = 0.0f;
    for (int d = 0; d < CDK; ++d)
      s += S.u.p5.qc[bl * 8 + m][h * 32 + d] * S.u.p5.kc[bl * 8 + n][h * 32 + d];
    S.u.p5.attn[bl][h][m][n] = s * 0.17677669529663687f;   // 1/sqrt(32)
  }
  __syncthreads();
  if (tid < 64) {
    int bl = tid >> 5, h = (tid >> 3) & 3, m = tid & 7;
    float mx = -1e30f;
    for (int n = 0; n < 8; ++n) mx = fmaxf(mx, S.u.p5.attn[bl][h][m][n]);
    float den = 0.0f;
    for (int n = 0; n < 8; ++n) { float e = expf(S.u.p5.attn[bl][h][m][n] - mx);
                                  S.u.p5.attn[bl][h][m][n] = e; den += e; }
    float inv = 1.0f / den;
    for (int n = 0; n < 8; ++n) S.u.p5.attn[bl][h][m][n] *= inv;
  }
  __syncthreads();
  for (int idx = tid; idx < 2048; idx += 256) {
    int r = idx >> 7, d = idx & 127, bl = r >> 3, m = r & 7, h = d >> 5;
    float s = 0.0f;
    for (int n = 0; n < 8; ++n) s += S.u.p5.attn[bl][h][m][n] * S.u.p5.vc[bl * 8 + n][d];
    S.outc[r][d] = f2bf(s);
  }
  __syncthreads();

  // phase 6 (WMMA+VALU): gated output, residual, mask blend, outputs
  for (int task = wid; task < 16; task += 8) {
    int mat = task >> 3, nt = task & 7;
    if (mat == 0)
      mm16(&S.outc[0][0], Wg_p, nt, 8, nullptr, nullptr, 0, 0,
           &S.u.p6.g1[0][0], 128, nt * 16, b_gate + nt * 16, nullptr, lane);
    else
      mm16(&S.outc[0][0], Wfc_p, nt, 8, nullptr, nullptr, 0, 0,
           &S.u.p6.g2[0][0], 128, nt * 16, b_fc + nt * 16, nullptr, lane);
  }
  __syncthreads();
  for (int idx = tid; idx < 2048; idx += 256) {
    int r = idx >> 7, d = idx & 127;
    float ao = sigm(S.u.p6.g1[r][d]) * tanhf(S.u.p6.g2[r][d]);
    float hn = S.Hnf[r][d] + ao;
    float mk = S.maskv[r];
    size_t g = (size_t)(b0 + (r >> 3)) * HID + (r & 7) * HSS + d;
    float ho = hx[g];
    out[OFF_HX + g] = mk * hn + (1.0f - mk) * ho;
    out[OFF_MF + g] = mk;
  }
  if (tid < 16) out[OFF_BM + (size_t)b0 * 8 + tid] = S.maskv[tid];
}

extern "C" void kernel_launch(void* const* d_in, const int* in_sizes, int n_in,
                              void* d_out, int out_size, void* d_ws, size_t ws_size,
                              hipStream_t stream) {
  const float* inp     = (const float*)d_in[0];
  const float* hx      = (const float*)d_in[1];
  const float* Wq_in   = (const float*)d_in[3];
  const float* Wk_in   = (const float*)d_in[4];
  const float* Wv_in   = (const float*)d_in[5];
  const float* W_ih    = (const float*)d_in[6];
  const float* b_ih    = (const float*)d_in[7];
  const float* W_hh    = (const float*)d_in[8];
  const float* b_hh    = (const float*)d_in[9];
  const float* W_read  = (const float*)d_in[10];
  const float* W_write = (const float*)d_in[11];
  const float* Wq_c    = (const float*)d_in[12];
  const float* bq_c    = (const float*)d_in[13];
  const float* Wk_c    = (const float*)d_in[14];
  const float* bk_c    = (const float*)d_in[15];
  const float* Wv_c    = (const float*)d_in[16];
  const float* bv_c    = (const float*)d_in[17];
  const float* W_fc    = (const float*)d_in[18];
  const float* b_fc    = (const float*)d_in[19];
  const float* W_gate  = (const float*)d_in[20];
  const float* b_gate  = (const float*)d_in[21];
  float* out = (float*)d_out;
  unsigned short* ws = (unsigned short*)d_ws;

  auto PK = [&](const float* s, size_t off, int K, int N, int mode) {
    int total = K * N;
    pack_weight<<<(total + 255) / 256, 256, 0, stream>>>(s, ws + off, K, N, mode);
  };
  PK(Wq_in,  PW_WQ,  128, 256, 0);
  PK(Wk_in,  PW_WK,  128, 256, 0);
  PK(Wv_in,  PW_WV,  128, 512, 0);
  PK(W_ih,   PW_WIH, 128, 1536, 1);
  PK(W_hh,   PW_WHH, 128, 1536, 1);
  PK(Wq_c,   PW_WQC, 128, 128, 0);
  PK(Wk_c,   PW_WKC, 128, 128, 0);
  PK(Wv_c,   PW_WVC, 128, 128, 0);
  PK(W_fc,   PW_WFC, 128, 128, 0);
  PK(W_gate, PW_WG,  128, 128, 0);
  PK(W_read, PW_WRD, 128, 16,  0);
  PK(W_write, PW_WWR, 128, 64, 2);

  scoff_cell_kernel<<<Bsz / 2, 256, sizeof(SMem), stream>>>(
      inp, hx, b_ih, b_hh, bq_c, bk_c, bv_c, b_fc, b_gate,
      ws + PW_WQ, ws + PW_WK, ws + PW_WV, ws + PW_WIH, ws + PW_WHH,
      ws + PW_WQC, ws + PW_WKC, ws + PW_WVC, ws + PW_WFC, ws + PW_WG,
      ws + PW_WRD, ws + PW_WWR, out);
}
